// GTN_hybrid_12687333392859
// MI455X (gfx1250) — matrix operand order repaired and checked by reading, not compiled, not run on hardware
//
#include <hip/hip_runtime.h>
#include <hip/hip_bf16.h>

typedef __attribute__((ext_vector_type(16))) _Float16 v16h;
typedef __attribute__((ext_vector_type(8)))  _Float16 v8h;
typedef __attribute__((ext_vector_type(8)))  float    v8f;

#define NN 20000      // nodes
#define NE 320000     // edges
#define DH 256        // hidden
#define NG 64         // graphs
#define NC 10         // classes
#define BN_EPS 1e-5f

// ---------- atomic helpers ----------
__device__ __forceinline__ void fatomic_add(float* p, float v) {
  __hip_atomic_fetch_add(p, v, __ATOMIC_RELAXED, __HIP_MEMORY_SCOPE_AGENT);
}
__device__ __forceinline__ void fatomic_max(float* addr, float value) {
  if (value >= 0.0f) atomicMax((int*)addr, __float_as_int(value));
  else               atomicMin((unsigned int*)addr, __float_as_uint(value));
}

// ---------- per-layer init: m=-inf, s=0, bn sums=0 ----------
__global__ void k_init_nodes(float* nm, float* ns, float* bnsum, float* bnsq) {
  int i = blockIdx.x * blockDim.x + threadIdx.x;
  if (i < NN) { nm[i] = -__builtin_inff(); ns[i] = 0.0f; }
  if (i < DH) { bnsum[i] = 0.0f; bnsq[i] = 0.0f; }
}

// ---------- f32 -> f16 activation convert (row-major, same layout) ----------
__global__ void k_cvt_act(const float* __restrict__ src, _Float16* __restrict__ dst,
                          int count) {
  int i = blockIdx.x * blockDim.x + threadIdx.x;
  if (i < count) dst[i] = (_Float16)src[i];
}

// ---------- weight convert + transpose: (4,K,DH) f32 -> (4,DH,K) f16 ----------
__global__ void k_cvt_wT(const float* __restrict__ W4, _Float16* __restrict__ Wt,
                         int Kdim) {
  int i = blockIdx.x * blockDim.x + threadIdx.x;
  int total = 4 * Kdim * DH;
  if (i >= total) return;
  int j = i / (Kdim * DH);
  int rem = i - j * Kdim * DH;
  int k = rem / DH;
  int c = rem - k * DH;
  Wt[(size_t)j * DH * Kdim + (size_t)c * Kdim + k] = (_Float16)W4[i];
}

// ---------- WMMA GEMM: out_j = A[NNxK] @ W[j] + b[j], j=0..3 ----------
// A16: f16 row-major [NN x KD]; Wt16: f16 col-major [4 x DH x KD]
// one wave computes 16 rows x 64 cols (4 accumulators, A fragment reused 4x)
// grid (NN/16, DH/64, 4), block 32
template <int KD>
__global__ void k_gemm4_wmma(const _Float16* __restrict__ A16,
                             const _Float16* __restrict__ Wt16,
                             const float* __restrict__ b4,
                             float* __restrict__ outQ, float* __restrict__ outK,
                             float* __restrict__ outV, float* __restrict__ outS) {
  const int lane = threadIdx.x & 31;
  const int half = lane >> 4;       // 0/1
  const int mn   = lane & 15;       // row (A) / col (B, C/D)
  const int row0 = blockIdx.x * 16;
  const int col0 = blockIdx.y * 64;
  const int j    = blockIdx.z;
  const float* __restrict__ bias = b4 + j * DH;
  float* __restrict__ out = (j == 0) ? outQ : (j == 1) ? outK : (j == 2) ? outV : outS;

  // A 16x32 f16 fragment: lane(half,m) holds 16 consecutive K at half*16
  const _Float16* Ar = A16 + (size_t)(row0 + mn) * KD + half * 16;
  // B 32x16 f16 fragment: lane(half,n) holds 16 consecutive K at half*16
  const _Float16* Wj = Wt16 + (size_t)j * DH * KD;
  const _Float16* B0 = Wj + (size_t)(col0 + 0  + mn) * KD + half * 16;
  const _Float16* B1 = Wj + (size_t)(col0 + 16 + mn) * KD + half * 16;
  const _Float16* B2 = Wj + (size_t)(col0 + 32 + mn) * KD + half * 16;
  const _Float16* B3 = Wj + (size_t)(col0 + 48 + mn) * KD + half * 16;

  v8f acc0 = {}, acc1 = {}, acc2 = {}, acc3 = {};
  for (int k0 = 0; k0 < KD; k0 += 32) {
    v8h alo = *(const v8h*)(Ar + k0);
    v8h ahi = *(const v8h*)(Ar + k0 + 8);
    v8h b0l = *(const v8h*)(B0 + k0), b0h = *(const v8h*)(B0 + k0 + 8);
    v8h b1l = *(const v8h*)(B1 + k0), b1h = *(const v8h*)(B1 + k0 + 8);
    v8h b2l = *(const v8h*)(B2 + k0), b2h = *(const v8h*)(B2 + k0 + 8);
    v8h b3l = *(const v8h*)(B3 + k0), b3h = *(const v8h*)(B3 + k0 + 8);
    v16h a, b0, b1, b2, b3;
#pragma unroll
    for (int i = 0; i < 8; ++i) {
      a[i] = alo[i];  a[i + 8] = ahi[i];
      b0[i] = b0l[i]; b0[i + 8] = b0h[i];
      b1[i] = b1l[i]; b1[i + 8] = b1h[i];
      b2[i] = b2l[i]; b2[i + 8] = b2h[i];
      b3[i] = b3l[i]; b3[i + 8] = b3h[i];
    }
    acc0 = __builtin_amdgcn_wmma_f32_16x16x32_f16(false, a, false, b0, (short)0, acc0, false, false);
    acc1 = __builtin_amdgcn_wmma_f32_16x16x32_f16(false, a, false, b1, (short)0, acc1, false, false);
    acc2 = __builtin_amdgcn_wmma_f32_16x16x32_f16(false, a, false, b2, (short)0, acc2, false, false);
    acc3 = __builtin_amdgcn_wmma_f32_16x16x32_f16(false, a, false, b3, (short)0, acc3, false, false);
  }

  // C/D layout: VGPR r -> row r + 8*half, col = lane&15
  float bb0 = bias[col0 + 0  + mn];
  float bb1 = bias[col0 + 16 + mn];
  float bb2 = bias[col0 + 32 + mn];
  float bb3 = bias[col0 + 48 + mn];
#pragma unroll
  for (int r = 0; r < 8; ++r) {
    size_t m = (size_t)(row0 + r + 8 * half) * DH;
    out[m + col0 + 0  + mn] = acc0[r] + bb0;
    out[m + col0 + 16 + mn] = acc1[r] + bb1;
    out[m + col0 + 32 + mn] = acc2[r] + bb2;
    out[m + col0 + 48 + mn] = acc3[r] + bb3;
  }
}

// ---------- edge logits: one wave per edge ----------
__global__ void k_edge_logits(const float* __restrict__ q, const float* __restrict__ k,
                              const int* __restrict__ src, const int* __restrict__ dst,
                              float* __restrict__ elog, float* __restrict__ nm) {
  int e = blockIdx.x * 8 + (threadIdx.x >> 5);
  int lane = threadIdx.x & 31;
  if (e >= NE) return;
  int s = src[e], d = dst[e];
  const float* qd = q + (size_t)d * DH;
  const float* ks = k + (size_t)s * DH;
  float sum = 0.0f;
#pragma unroll
  for (int i = 0; i < 8; ++i) {
    int c = lane + i * 32;
    sum += qd[c] * ks[c];
  }
#pragma unroll
  for (int off = 16; off > 0; off >>= 1) sum += __shfl_xor(sum, off, 32);
  if (lane == 0) {
    float lg = sum * 0.0625f;             // 1/sqrt(256)
    elog[e] = lg;
    fatomic_max(&nm[d], lg);
  }
}

// ---------- edge exp + segment sum ----------
__global__ void k_edge_exp(const float* __restrict__ elog, const float* __restrict__ nm,
                           const int* __restrict__ dst,
                           float* __restrict__ ea, float* __restrict__ ns) {
  int e = blockIdx.x * blockDim.x + threadIdx.x;
  if (e >= NE) return;
  int d = dst[e];
  float a = __expf(elog[e] - nm[d]);
  ea[e] = a;
  fatomic_add(&ns[d], a);
}

// ---------- weighted aggregation: out[dst] += alpha * v[src]; one wave/edge ----------
__global__ void k_edge_aggregate(const float* __restrict__ ea, const float* __restrict__ ns,
                                 const float* __restrict__ v,
                                 const int* __restrict__ src, const int* __restrict__ dst,
                                 float* __restrict__ out) {
  int e = blockIdx.x * 8 + (threadIdx.x >> 5);
  int lane = threadIdx.x & 31;
  if (e >= NE) return;
  int s = src[e], d = dst[e];
  float alpha = ea[e] / (ns[d] + 1e-16f);
  const float* vs = v + (size_t)s * DH;
  float* od = out + (size_t)d * DH;
#pragma unroll
  for (int i = 0; i < 8; ++i) {
    int c = lane + i * 32;
    fatomic_add(&od[c], alpha * vs[c]);
  }
}

// ---------- BN stats: 250 blocks x 256 threads, 80 rows each ----------
__global__ void k_bn_stats(const float* __restrict__ h, float* __restrict__ bnsum,
                           float* __restrict__ bnsq) {
  int c = threadIdx.x;
  int r0 = blockIdx.x * 80;
  float s = 0.0f, s2 = 0.0f;
  for (int r = 0; r < 80; ++r) {
    float x = h[(size_t)(r0 + r) * DH + c];
    s += x; s2 += x * x;
  }
  fatomic_add(&bnsum[c], s);
  fatomic_add(&bnsq[c], s2);
}

__global__ void k_bn_finalize(const float* __restrict__ bnsum, const float* __restrict__ bnsq,
                              const float* __restrict__ gamma, const float* __restrict__ beta,
                              float* __restrict__ scale, float* __restrict__ shift) {
  int c = threadIdx.x;
  const float invN = 1.0f / (float)NN;
  float mu  = bnsum[c] * invN;
  float var = bnsq[c] * invN - mu * mu;
  float sc  = gamma[c] * rsqrtf(var + BN_EPS);
  scale[c] = sc;
  shift[c] = beta[c] - mu * sc;
}

__global__ void k_bn_apply_relu(float* __restrict__ h, const float* __restrict__ scale,
                                const float* __restrict__ shift) {
  size_t idx = (size_t)blockIdx.x * blockDim.x + threadIdx.x;
  int c = threadIdx.x;  // blockDim == DH
  float y = h[idx] * scale[c] + shift[c];
  h[idx] = fmaxf(y, 0.0f);
}

// ---------- pooling ----------
__global__ void k_pool_init(float* __restrict__ pool_and_cnt) {
  int i = blockIdx.x * blockDim.x + threadIdx.x;
  if (i < NG * DH + NG) pool_and_cnt[i] = 0.0f;
}

__global__ void k_pool(const float* __restrict__ h, const int* __restrict__ batch,
                       float* __restrict__ pool, float* __restrict__ cnt) {
  int n = blockIdx.x;
  int c = threadIdx.x;
  int g = batch[n];
  fatomic_add(&pool[(size_t)g * DH + c], h[(size_t)n * DH + c]);
  if (c == 0) fatomic_add(&cnt[g], 1.0f);
}

__global__ void k_final_linear(const float* __restrict__ pool, const float* __restrict__ cnt,
                               const float* __restrict__ linW, const float* __restrict__ linb,
                               float* __restrict__ out) {
  int idx = blockIdx.x * blockDim.x + threadIdx.x;
  if (idx >= NG * NC) return;
  int g = idx / NC, cls = idx % NC;
  float inv = 1.0f / fmaxf(cnt[g], 1.0f);
  float s = linb[cls];
  const float* pg = pool + (size_t)g * DH;
  for (int c = 0; c < DH; ++c) s += pg[c] * inv * linW[c * NC + cls];
  out[idx] = s;
}

// ---------------- host orchestration ----------------
extern "C" void kernel_launch(void* const* d_in, const int* in_sizes, int n_in,
                              void* d_out, int out_size, void* d_ws, size_t ws_size,
                              hipStream_t stream) {
  (void)in_sizes; (void)n_in; (void)out_size; (void)ws_size;
  const float* x       = (const float*)d_in[0];
  const int*   e0      = (const int*)d_in[1];   // (2, NE)
  const int*   e1      = (const int*)d_in[2];   // (2, NE)
  const int*   batch   = (const int*)d_in[3];
  const float* conv1_W = (const float*)d_in[4]; // (4,128,256)
  const float* conv1_b = (const float*)d_in[5]; // (4,256)
  const float* convs_W = (const float*)d_in[6]; // (5,4,256,256)
  const float* convs_b = (const float*)d_in[7]; // (5,4,256)
  const float* bn_g    = (const float*)d_in[8]; // (6,256)
  const float* bn_b    = (const float*)d_in[9]; // (6,256)
  const float* lin_W   = (const float*)d_in[10];// (256,10)
  const float* lin_b   = (const float*)d_in[11];// (10)
  float* out = (float*)d_out;

  // workspace layout (float-sized units; all chunks multiple of 4 floats -> 16B aligned)
  float* ws = (float*)d_ws;
  size_t off = 0;
  float* bufA  = ws + off; off += (size_t)NN * DH;
  float* bufB  = ws + off; off += (size_t)NN * DH;
  float* q     = ws + off; off += (size_t)NN * DH;
  float* kbuf  = ws + off; off += (size_t)NN * DH;
  float* v     = ws + off; off += (size_t)NN * DH;
  _Float16* A16 = (_Float16*)(ws + off); off += (size_t)NN * DH / 2;
  _Float16* Wt16 = (_Float16*)(ws + off); off += (size_t)4 * DH * DH / 2;
  float* elog  = ws + off; off += NE;
  float* ea    = ws + off; off += NE;
  float* nm    = ws + off; off += NN;
  float* ns    = ws + off; off += NN;
  float* bnsum = ws + off; off += DH;
  float* bnsq  = ws + off; off += DH;
  float* bnsc  = ws + off; off += DH;
  float* bnsh  = ws + off; off += DH;
  float* pool  = ws + off; off += (size_t)NG * DH;
  float* cnt   = ws + off; off += NG;

  const float* hin = x;
  for (int l = 0; l < 6; ++l) {
    // edge set: layer 0 -> edge0; layers 1,2 -> edge1; layers 3,4,5 -> edge0
    const int* src = (l == 1 || l == 2) ? (e1)      : (e0);
    const int* dst = (l == 1 || l == 2) ? (e1 + NE) : (e0 + NE);
    const float* Wl; const float* bl; int Kdim;
    if (l == 0) { Wl = conv1_W; bl = conv1_b; Kdim = 128; }
    else {
      Wl = convs_W + (size_t)(l - 1) * 4 * DH * DH;
      bl = convs_b + (size_t)(l - 1) * 4 * DH;
      Kdim = DH;
    }
    float* hout = (l % 2 == 0) ? bufA : bufB;

    k_init_nodes<<<(NN + 255) / 256, 256, 0, stream>>>(nm, ns, bnsum, bnsq);
    // precision pre-passes: convert activations + transpose/convert weights
    k_cvt_act<<<((size_t)NN * Kdim + 255) / 256, 256, 0, stream>>>(hin, A16, NN * Kdim);
    k_cvt_wT<<<(4 * Kdim * DH + 255) / 256, 256, 0, stream>>>(Wl, Wt16, Kdim);
    if (Kdim == 128)
      k_gemm4_wmma<128><<<dim3(NN / 16, DH / 64, 4), 32, 0, stream>>>(
          A16, Wt16, bl, q, kbuf, v, hout);
    else
      k_gemm4_wmma<256><<<dim3(NN / 16, DH / 64, 4), 32, 0, stream>>>(
          A16, Wt16, bl, q, kbuf, v, hout);
    k_edge_logits<<<NE / 8, 256, 0, stream>>>(q, kbuf, src, dst, elog, nm);
    k_edge_exp<<<NE / 256, 256, 0, stream>>>(elog, nm, dst, ea, ns);
    k_edge_aggregate<<<NE / 8, 256, 0, stream>>>(ea, ns, v, src, dst, hout);
    k_bn_stats<<<NN / 80, 256, 0, stream>>>(hout, bnsum, bnsq);
    k_bn_finalize<<<1, 256, 0, stream>>>(bnsum, bnsq, bn_g + l * DH, bn_b + l * DH,
                                         bnsc, bnsh);
    k_bn_apply_relu<<<NN, 256, 0, stream>>>(hout, bnsc, bnsh);
    hin = hout;
  }

  k_pool_init<<<(NG * DH + NG + 255) / 256, 256, 0, stream>>>(pool);
  k_pool<<<NN, 256, 0, stream>>>(hin, batch, pool, cnt);
  k_final_linear<<<(NG * NC + 255) / 256, 256, 0, stream>>>(pool, cnt, lin_W, lin_b, out);
}
